// BITModel_27075473834144
// MI455X (gfx1250) — matrix operand order repaired and checked by reading, not compiled, not run on hardware
//
#include <hip/hip_runtime.h>
#include <hip/hip_bf16.h>
#include <math.h>

#define Bz 16
#define Nn 4096
#define Dd 512
#define Hh 8
#define Ee 64
#define Mm 4
#define SCALE_QK 0.125f  // 1/sqrt(64)

typedef __attribute__((ext_vector_type(16))) __bf16 v16bf;
typedef __attribute__((ext_vector_type(8)))  float  v8f;
typedef __attribute__((ext_vector_type(4)))  unsigned uvec4;
typedef __attribute__((ext_vector_type(4)))  float  fvec4;

union BV { uvec4 q[2]; unsigned u[8]; v16bf v; };

__device__ __forceinline__ unsigned f2bf(float f) {
    unsigned u = __builtin_bit_cast(unsigned, f);
    return (u + 0x7FFFu + ((u >> 16) & 1u)) >> 16;   // RNE fp32 -> bf16
}
__device__ __forceinline__ unsigned pk2(float a, float b) {
    return f2bf(a) | (f2bf(b) << 16);
}
__device__ __forceinline__ float bf2f(unsigned short h) {
    return __builtin_bit_cast(float, ((unsigned)h) << 16);
}

// ---- convert x (fp32) -> xbf (bf16), 8 elements/thread, fully coalesced ----
__global__ void __launch_bounds__(256) k_xbf(const float* __restrict__ x,
                                             unsigned short* __restrict__ xbf) {
    size_t base = ((size_t)blockIdx.x * 256 + threadIdx.x) * 8;
    fvec4 f0 = *(const fvec4*)(x + base);
    fvec4 f1 = *(const fvec4*)(x + base + 4);
    uvec4 o;
    o[0] = pk2(f0[0], f0[1]); o[1] = pk2(f0[2], f0[3]);
    o[2] = pk2(f1[0], f1[1]); o[3] = pk2(f1[2], f1[3]);
    *(uvec4*)(xbf + base) = o;
}

// ---- transpose/convert Wk|Wv (512x512 each) -> Wt bf16 (1024 cols x 512 k) ----
__global__ void k_wt(const float* __restrict__ Wk, const float* __restrict__ Wv,
                     unsigned short* __restrict__ Wt) {
    int idx = blockIdx.x * 256 + threadIdx.x;       // 0..524287
    int c = idx >> 9, k = idx & 511;
    float v = (c < 512) ? Wk[(size_t)k * 512 + c] : Wv[(size_t)k * 512 + (c - 512)];
    Wt[(size_t)c * 512 + k] = (unsigned short)f2bf(v);
}

// ---- Q0 = pattern @ Wq + bq : (4, 512) fp32 ----
__global__ void k_qinit(const float* __restrict__ pattern, const float* __restrict__ Wq,
                        const float* __restrict__ bq, float* __restrict__ Qinit) {
    int idx = blockIdx.x * 256 + threadIdx.x;       // 0..2047
    int m = idx >> 9, c = idx & 511;
    float acc = bq[c];
    const float* pr = pattern + m * 512;
    for (int k = 0; k < 512; ++k) acc = fmaf(pr[k], Wq[(size_t)k * 512 + c], acc);
    Qinit[idx] = acc;
}

// ---- K = x@Wk+bk, V = x@Wv+bv via bf16 WMMA; out bf16 [b][h][n][e] ----
// Each wave computes a 16x32 output tile: one A load feeds two WMMAs
// (3 b128 loads per WMMA instead of 4).
__global__ void __launch_bounds__(256) k_proj(const unsigned short* __restrict__ xbf,
        const unsigned short* __restrict__ Wt,
        const float* __restrict__ bk, const float* __restrict__ bv,
        unsigned short* __restrict__ Kbf, unsigned short* __restrict__ Vbf) {
    int wave = threadIdx.x >> 5, lane = threadIdx.x & 31;
    int rtile = blockIdx.x;                 // 0..4095 (16 rows each of B*N)
    int pair  = blockIdx.y * 8 + wave;      // 0..31  (32 cols each of 1024)
    int mlane = lane & 15, kh = lane >> 4;
    size_t row = (size_t)rtile * 16 + mlane;
    const unsigned short* xrow = xbf + row * Dd + kh * 16;  // A: elems 0..7 at +0, 8..15 at +8
    int col0 = pair * 32 + mlane;           // B col for first tile
    int col1 = col0 + 16;                   // B col for second tile
    const unsigned short* wrow0 = Wt + (size_t)col0 * Dd + kh * 16;
    const unsigned short* wrow1 = Wt + (size_t)col1 * Dd + kh * 16;

    v8f acc0 = {}, acc1 = {};
    for (int ks = 0; ks < 16; ++ks) {
        int kb = ks * 32;
        BV a, b0, b1;
        a.q[0] = *(const uvec4*)(xrow + kb);        // K = kb + kh*16 + 0..7
        a.q[1] = *(const uvec4*)(xrow + kb + 8);    // K = kb + kh*16 + 8..15
        const uvec4* wp0 = (const uvec4*)(wrow0 + kb);
        b0.q[0] = wp0[0]; b0.q[1] = wp0[1];         // K = kb + kh*16 + 0..15
        const uvec4* wp1 = (const uvec4*)(wrow1 + kb);
        b1.q[0] = wp1[0]; b1.q[1] = wp1[1];
        acc0 = __builtin_amdgcn_wmma_f32_16x16x32_bf16(false, a.v, false, b0.v,
                                                       (short)0, acc0, false, false);
        acc1 = __builtin_amdgcn_wmma_f32_16x16x32_bf16(false, a.v, false, b1.v,
                                                       (short)0, acc1, false, false);
    }
    for (int half = 0; half < 2; ++half) {
        int col = half ? col1 : col0;
        v8f acc = half ? acc1 : acc0;
        float bias = (col < 512) ? bk[col] : bv[col - 512];
        int hK = ((col < 512) ? col : (col - 512)) >> 6;
        int e = col & 63;
        unsigned short* dst = (col < 512) ? Kbf : Vbf;
        for (int r = 0; r < 8; ++r) {
            int mm = r + 8 * kh;
            size_t grow = (size_t)rtile * 16 + mm;
            int bb = (int)(grow >> 12), nn = (int)(grow & 4095);
            float v = acc[r] + bias;
            dst[(((size_t)bb * Hh + hK) * Nn + nn) * 64 + e] = (unsigned short)f2bf(v);
        }
    }
}

// ---- fused Hopfield iterations per (b,h): WMMA scores + sparsemax + updates ----
__global__ void __launch_bounds__(256) k_hopfield(const float* __restrict__ Qinit,
        const unsigned short* __restrict__ Kbf, const unsigned short* __restrict__ Vbf,
        float* __restrict__ A_ws, float* __restrict__ Hout) {
    __shared__ unsigned short Qb[Mm * Ee];
    __shared__ float red[256];
    int bh = blockIdx.x;                    // 0..127
    int b = bh >> 3, h = bh & 7;
    int t = threadIdx.x;
    int wave = t >> 5, lane = t & 31;
    int mlane = lane & 15, kh = lane >> 4;

    {   // load Q0 slice for this head, convert to bf16
        int m = t >> 6, e = t & 63;
        Qb[t] = (unsigned short)f2bf(Qinit[m * 512 + h * 64 + e]);
    }
    __syncthreads();

    const unsigned short* Kslice = Kbf + (size_t)bh * Nn * 64;
    const unsigned short* Vslice = Vbf + (size_t)bh * Nn * 64;
    float* Arow0 = A_ws + (size_t)bh * Mm * Nn;

    for (int step = 0; step < 3; ++step) {
        // (a) scores = Q K^T * scale  (WMMA, rows 4..15 of Q zero)
        BV a0, a1;
        for (int j = 0; j < 4; ++j) {
            int k = kh * 8 + 2 * j;
            unsigned q00 = 0, q01 = 0, q10 = 0, q11 = 0;
            unsigned q20 = 0, q21 = 0, q30 = 0, q31 = 0;
            if (mlane < Mm) {
                const unsigned short* qm = Qb + mlane * 64;
                q00 = qm[k];      q01 = qm[k + 1];
                q10 = qm[k + 16]; q11 = qm[k + 17];
                q20 = qm[k + 32]; q21 = qm[k + 33];
                q30 = qm[k + 48]; q31 = qm[k + 49];
            }
            a0.u[j]     = q00 | (q01 << 16);
            a0.u[4 + j] = q10 | (q11 << 16);
            a1.u[j]     = q20 | (q21 << 16);
            a1.u[4 + j] = q30 | (q31 << 16);
        }
        for (int nt = wave; nt < Nn / 16; nt += 8) {
            const unsigned short* kp = Kslice + (size_t)(nt * 16 + mlane) * 64 + kh * 8;
            BV b0, b1;
            const uvec4* p0 = (const uvec4*)kp;
            b0.q[0] = p0[0]; b0.q[1] = p0[2];
            const uvec4* p1 = (const uvec4*)(kp + 32);
            b1.q[0] = p1[0]; b1.q[1] = p1[2];
            v8f acc = {};
            acc = __builtin_amdgcn_wmma_f32_16x16x32_bf16(false, a0.v, false, b0.v,
                                                          (short)0, acc, false, false);
            acc = __builtin_amdgcn_wmma_f32_16x16x32_bf16(false, a1.v, false, b1.v,
                                                          (short)0, acc, false, false);
            if (lane < 16) {
                for (int r = 0; r < Mm; ++r)
                    Arow0[(size_t)r * Nn + nt * 16 + lane] = acc[r] * SCALE_QK;
            }
        }
        __threadfence();
        __syncthreads();

        // (b) sparsemax along N (exact simplex projection, Michelot iteration)
        for (int m = 0; m < Mm; ++m) {
            float* z = Arow0 + (size_t)m * Nn;
            float ps = 0.f;
            for (int i = t; i < Nn; i += 256) ps += z[i];
            red[t] = ps; __syncthreads();
            for (int s = 128; s > 0; s >>= 1) { if (t < s) red[t] += red[t + s]; __syncthreads(); }
            float tau = (red[0] - 1.0f) / (float)Nn;
            __syncthreads();
            for (int it = 0; it < 16; ++it) {
                float s1 = 0.f, c1 = 0.f;
                for (int i = t; i < Nn; i += 256) {
                    float zi = z[i];
                    if (zi > tau) { s1 += zi; c1 += 1.0f; }
                }
                red[t] = s1; __syncthreads();
                for (int s = 128; s > 0; s >>= 1) { if (t < s) red[t] += red[t + s]; __syncthreads(); }
                float S = red[0]; __syncthreads();
                red[t] = c1; __syncthreads();
                for (int s = 128; s > 0; s >>= 1) { if (t < s) red[t] += red[t + s]; __syncthreads(); }
                float C = red[0]; __syncthreads();
                tau = (S - 1.0f) / fmaxf(C, 1.0f);
            }
            for (int i = t; i < Nn; i += 256) z[i] = fmaxf(z[i] - tau, 0.0f);
            __syncthreads();
        }
        __threadfence();
        __syncthreads();

        // (c) Q <- A K (steps 0,1)  or  out <- A V (step 2); M*E == 256 threads
        int m = t >> 6, e = t & 63;
        const float* ar = Arow0 + (size_t)m * Nn;
        if (step < 2) {
            const unsigned short* kc = Kslice + e;
            float acc = 0.f;
            for (int n = 0; n < Nn; n += 4) {
                acc = fmaf(ar[n],     bf2f(kc[(size_t)(n)     * 64]), acc);
                acc = fmaf(ar[n + 1], bf2f(kc[(size_t)(n + 1) * 64]), acc);
                acc = fmaf(ar[n + 2], bf2f(kc[(size_t)(n + 2) * 64]), acc);
                acc = fmaf(ar[n + 3], bf2f(kc[(size_t)(n + 3) * 64]), acc);
            }
            __syncthreads();
            Qb[t] = (unsigned short)f2bf(acc);
            __syncthreads();
        } else {
            const unsigned short* vc = Vslice + e;
            float acc = 0.f;
            for (int n = 0; n < Nn; n += 4) {
                acc = fmaf(ar[n],     bf2f(vc[(size_t)(n)     * 64]), acc);
                acc = fmaf(ar[n + 1], bf2f(vc[(size_t)(n + 1) * 64]), acc);
                acc = fmaf(ar[n + 2], bf2f(vc[(size_t)(n + 2) * 64]), acc);
                acc = fmaf(ar[n + 3], bf2f(vc[(size_t)(n + 3) * 64]), acc);
            }
            Hout[((size_t)b * Mm + m) * 512 + h * 64 + e] = acc;
        }
    }
}

// ---- out2 = gelu(Hout @ Wo + bo); y[b] = sum(out2 * Wf) + bf ----
__global__ void __launch_bounds__(256) k_head(const float* __restrict__ Hout,
        const float* __restrict__ Wo, const float* __restrict__ bo,
        const float* __restrict__ Wf, const float* __restrict__ bfin,
        float* __restrict__ out) {
    __shared__ float hl[Mm * 512];
    __shared__ float red[256];
    int b = blockIdx.x, t = threadIdx.x;
    for (int i = t; i < Mm * 512; i += 256) hl[i] = Hout[(size_t)b * Mm * 512 + i];
    __syncthreads();
    float part = 0.f;
    for (int idx = t; idx < Mm * 512; idx += 256) {
        int m = idx >> 9, d = idx & 511;
        float acc = bo[d];
        const float* hr = hl + m * 512;
        for (int k = 0; k < 512; ++k) acc = fmaf(hr[k], Wo[(size_t)k * 512 + d], acc);
        float g = 0.5f * acc * (1.0f + erff(acc * 0.70710678118654752f)); // exact gelu
        part = fmaf(g, Wf[idx], part);
    }
    red[t] = part; __syncthreads();
    for (int s = 128; s > 0; s >>= 1) { if (t < s) red[t] += red[t + s]; __syncthreads(); }
    if (t == 0) out[b] = red[0] + bfin[0];
}

extern "C" void kernel_launch(void* const* d_in, const int* in_sizes, int n_in,
                              void* d_out, int out_size, void* d_ws, size_t ws_size,
                              hipStream_t stream) {
    (void)in_sizes; (void)n_in; (void)out_size; (void)ws_size;
    const float* x       = (const float*)d_in[0];
    const float* pattern = (const float*)d_in[1];
    const float* Wq      = (const float*)d_in[2];
    const float* bq      = (const float*)d_in[3];
    const float* Wk      = (const float*)d_in[4];
    const float* bk      = (const float*)d_in[5];
    const float* Wv      = (const float*)d_in[6];
    const float* bv      = (const float*)d_in[7];
    const float* Wo      = (const float*)d_in[8];
    const float* bo      = (const float*)d_in[9];
    const float* Wf      = (const float*)d_in[10];
    const float* bff     = (const float*)d_in[11];
    float* out = (float*)d_out;

    char* ws = (char*)d_ws;
    size_t off = 0;
    auto take = [&](size_t bytes) -> char* {
        char* p = ws + off;
        off = (off + bytes + 255) & ~(size_t)255;
        return p;
    };
    unsigned short* xbf   = (unsigned short*)take((size_t)Bz * Nn * Dd * 2);       // 64 MB
    unsigned short* Wt    = (unsigned short*)take((size_t)1024 * 512 * 2);         // 1 MB
    float*          Qinit = (float*)take((size_t)Mm * 512 * 4);                    // 8 KB
    unsigned short* Kbf   = (unsigned short*)take((size_t)Bz * Hh * Nn * 64 * 2);  // 64 MB
    unsigned short* Vbf   = (unsigned short*)take((size_t)Bz * Hh * Nn * 64 * 2);  // 64 MB
    float*          A_ws  = (float*)take((size_t)Bz * Hh * Mm * Nn * 4);           // 8 MB
    float*          Hout  = (float*)take((size_t)Bz * Mm * 512 * 4);               // 128 KB

    k_xbf    <<<(Bz * Nn * Dd) / (256 * 8), 256, 0, stream>>>(x, xbf);
    k_wt     <<<2048, 256, 0, stream>>>(Wk, Wv, Wt);
    k_qinit  <<<8,    256, 0, stream>>>(pattern, Wq, bq, Qinit);
    k_proj   <<<dim3(4096, 4), 256, 0, stream>>>(xbf, Wt, bk, bv, Kbf, Vbf);
    k_hopfield<<<Bz * Hh, 256, 0, stream>>>(Qinit, Kbf, Vbf, A_ws, Hout);
    k_head   <<<Bz,   256, 0, stream>>>(Hout, Wo, bo, Wf, bff, out);
}
